// GNN_23983097381472
// MI455X (gfx1250) — matrix operand order repaired and checked by reading, not compiled, not run on hardware
//
#include <hip/hip_runtime.h>
#include <hip/hip_bf16.h>

// ---------------------------------------------------------------------------
// GNN (768 nodes, H=100, 3 iterations), fp32 I/O, f16 WMMA for edge GEMMs.
//
// Key algebraic rewrite:  (e @ W_dv1 + b).reshape(n,n,H).sum(axis=1)
//                       = (sum_j e[i,j,:]) @ W_dv1 + n*b
// so the only O(N^2 * H * H) work left is the two edge GEMMs, fused into one
// kernel that also accumulates the per-node row sums (E_sum) with atomics.
// ---------------------------------------------------------------------------

#define NN      768          // atoms
#define HD      100          // hidden
#define HP      112          // padded output cols  (7 * 16)
#define KP      128          // padded K            (4 * 32)
#define NEDGE   (NN * NN)    // 589824 edge rows
#define WPB     4            // waves per block in edge kernel

typedef __attribute__((ext_vector_type(16))) _Float16 v16h;
typedef __attribute__((ext_vector_type(8)))  _Float16 v8h;
typedef __attribute__((ext_vector_type(8)))  float    v8f;

__device__ __forceinline__ v16h cat16(v8h lo, v8h hi) {
    return __builtin_shufflevector(lo, hi, 0,1,2,3,4,5,6,7,8,9,10,11,12,13,14,15);
}

// ---------------------------------------------------------------------------
// Fused edge update (the hot kernel): per 16-row tile of e (each tile shares
// node index i since 768 % 16 == 0):
//   f   = relu(e @ W_eij + b_eij + dots[row])       (f16 WMMA, f32 acc)
//   e  += f @ W_de + b_de                            (fp32 residual in global)
//   E_sum[i,:] += sum over the 16 rows of new e      (atomicAdd)
// ---------------------------------------------------------------------------
__global__ __launch_bounds__(WPB * 32) void edge_update_kernel(
    float* __restrict__ e,
    const float* __restrict__ W_eij, const float* __restrict__ b_eij,
    const float* __restrict__ W_de,  const float* __restrict__ b_de,
    const float* __restrict__ Dots,
    float* __restrict__ E_sum)
{
    __shared__ _Float16 sW[HP * KP];            // weights, transposed [n][k]
    __shared__ _Float16 sA[WPB][16 * KP];       // per-wave e tile (f16, padded)
    __shared__ _Float16 sF[WPB][16 * KP];       // per-wave f tile (f16, padded)
    __shared__ float    sDot[WPB][16];

    const int tid   = threadIdx.x;
    const int wave  = tid >> 5;
    const int lane  = tid & 31;
    const int khalf = lane >> 4;                // 0 or 1
    const int lmod  = lane & 15;
    const size_t r0 = ((size_t)blockIdx.x * WPB + wave) * 16;   // first edge row

    // ---- stage W_eij transposed (zero-padded to 112x128) ----
    for (int idx = tid; idx < HP * KP; idx += WPB * 32) {
        const int n = idx >> 7;                 // / KP
        const int k = idx & (KP - 1);
        const float w = (n < HD && k < HD) ? W_eij[k * HD + n] : 0.f;
        sW[idx] = (_Float16)w;
    }
    // ---- stage e tile: fp32 -> f16, K padded to 128 ----
    {
        const float* erow0 = e + r0 * HD;
        for (int t = lane; t < 16 * 25; t += 32) {       // 25 float4 per row
            const int row = t / 25, q = t - row * 25;
            const float4 v = ((const float4*)(erow0 + (size_t)row * HD))[q];
            _Float16* dst = &sA[wave][row * KP + q * 4];
            dst[0] = (_Float16)v.x; dst[1] = (_Float16)v.y;
            dst[2] = (_Float16)v.z; dst[3] = (_Float16)v.w;
        }
        for (int t = lane; t < 16 * 7; t += 32) {        // zero pad k = 100..127
            const int row = t / 7, q = t - row * 7;
            _Float16* dst = &sA[wave][row * KP + HD + q * 4];
            dst[0] = dst[1] = dst[2] = dst[3] = (_Float16)0.f;
        }
        if (lane < 16) sDot[wave][lane] = Dots[r0 + lane];
    }
    __syncthreads();

    // ---- GEMM1: f = relu(e @ W_eij + b_eij + dots) ----
    const _Float16* myA = sA[wave];
    for (int nt = 0; nt < 7; ++nt) {
        v8f acc = {};
        const int n = nt * 16 + lmod;
        #pragma unroll
        for (int kst = 0; kst < 4; ++kst) {
            const int kb = kst * 32;
            // A frag: lane holds row M=lmod; K 0..15 / 16..31 (same order as B)
            const v8h alo = *(const v8h*)&myA[lmod * KP + kb + khalf * 16];
            const v8h ahi = *(const v8h*)&myA[lmod * KP + kb + khalf * 16 + 8];
            // B frag (from transposed weights): lane holds col n; K 0..15 / 16..31
            const v8h blo = *(const v8h*)&sW[n * KP + kb + khalf * 16];
            const v8h bhi = *(const v8h*)&sW[n * KP + kb + khalf * 16 + 8];
            acc = __builtin_amdgcn_wmma_f32_16x16x32_f16(
                false, cat16(alo, ahi), false, cat16(blo, bhi),
                (short)0, acc, false, false);
        }
        const float bn = (n < HD) ? b_eij[n] : 0.f;
        #pragma unroll
        for (int r = 0; r < 8; ++r) {
            const int m = khalf * 8 + r;                // C/D layout row
            float f = acc[r] + bn + sDot[wave][m];
            f = (n < HD) ? fmaxf(f, 0.f) : 0.f;         // padded cols must be 0
            sF[wave][m * KP + n] = (_Float16)f;
        }
    }
    // zero-pad sF columns 112..127 (K padding for GEMM2)
    for (int t = lane; t < 16 * 4; t += 32) {
        const int row = t >> 2, q = t & 3;
        _Float16* dst = &sF[wave][row * KP + HP + q * 4];
        dst[0] = dst[1] = dst[2] = dst[3] = (_Float16)0.f;
    }
    __syncthreads();

    // ---- restage weights: W_de transposed ----
    for (int idx = tid; idx < HP * KP; idx += WPB * 32) {
        const int n = idx >> 7;
        const int k = idx & (KP - 1);
        const float w = (n < HD && k < HD) ? W_de[k * HD + n] : 0.f;
        sW[idx] = (_Float16)w;
    }
    __syncthreads();

    // ---- GEMM2 + fp32 residual + fused row-sum into E_sum ----
    const int i_node = (int)(r0 / NN);                  // shared by all 16 rows
    const _Float16* myF = sF[wave];
    for (int nt = 0; nt < 7; ++nt) {
        v8f acc = {};
        const int n = nt * 16 + lmod;
        #pragma unroll
        for (int kst = 0; kst < 4; ++kst) {
            const int kb = kst * 32;
            const v8h alo = *(const v8h*)&myF[lmod * KP + kb + khalf * 16];
            const v8h ahi = *(const v8h*)&myF[lmod * KP + kb + khalf * 16 + 8];
            const v8h blo = *(const v8h*)&sW[n * KP + kb + khalf * 16];
            const v8h bhi = *(const v8h*)&sW[n * KP + kb + khalf * 16 + 8];
            acc = __builtin_amdgcn_wmma_f32_16x16x32_f16(
                false, cat16(alo, ahi), false, cat16(blo, bhi),
                (short)0, acc, false, false);
        }
        if (n < HD) {
            const float bd = b_de[n];
            float rsum = 0.f;
            #pragma unroll
            for (int r = 0; r < 8; ++r) {
                const int m = khalf * 8 + r;
                float* ep = e + (r0 + m) * HD + n;      // fp32 residual (cached)
                const float enew = *ep + acc[r] + bd;
                *ep = enew;
                rsum += enew;
            }
            rsum += __shfl_down(rsum, 16);              // fold the two M-halves
            if (lane < 16) atomicAdd(&E_sum[(size_t)i_node * HD + n], rsum);
        }
    }
}

// ---------------------------------------------------------------------------
// Small kernels (≪1% of the FLOPs; plain VALU is fine)
// ---------------------------------------------------------------------------
__global__ void init_h_kernel(const float* __restrict__ nf,
                              const float* __restrict__ W,
                              const float* __restrict__ b,
                              float* __restrict__ h) {
    const int idx = blockIdx.x * blockDim.x + threadIdx.x;
    if (idx >= NN * HD) return;
    const int i = idx / HD, c = idx - i * HD;
    float acc = b[c];
    const float* row = nf + i * 32;
    #pragma unroll
    for (int k = 0; k < 32; ++k) acc += row[k] * W[k * HD + c];
    h[idx] = acc;
}

__global__ void init_e_kernel(const float* __restrict__ ea,
                              const float* __restrict__ W,
                              const float* __restrict__ b,
                              float* __restrict__ e) {
    const int idx = blockIdx.x * blockDim.x + threadIdx.x;
    if (idx >= NEDGE * HD) return;
    const int row = idx / HD, c = idx - row * HD;
    float acc = b[c];
    const float* r = ea + (size_t)row * 16;
    #pragma unroll
    for (int k = 0; k < 16; ++k) acc += r[k] * W[k * HD + c];
    e[idx] = acc;
}

__global__ void fifj_kernel(const float* __restrict__ h,
                            const float* __restrict__ W_i,
                            const float* __restrict__ W_j,
                            float* __restrict__ F_i,
                            float* __restrict__ F_j) {
    const int idx = blockIdx.x * blockDim.x + threadIdx.x;
    if (idx >= NN * HD) return;
    const int i = idx / HD, c = idx - i * HD;
    float ai = 0.f, aj = 0.f;
    const float* hr = h + i * HD;
    for (int k = 0; k < HD; ++k) {
        const float hv = hr[k];
        ai += hv * W_i[k * HD + c];
        aj += hv * W_j[k * HD + c];
    }
    F_i[idx] = ai;
    F_j[idx] = aj;
}

__global__ void dots_kernel(const float* __restrict__ F_i,
                            const float* __restrict__ F_j,
                            float* __restrict__ Dots) {
    const int idx = blockIdx.x * blockDim.x + threadIdx.x;
    if (idx >= NEDGE) return;
    const int i = idx / NN, j = idx - i * NN;
    const float* a = F_i + i * HD;
    const float* b = F_j + j * HD;
    float acc = 0.f;
    for (int k = 0; k < HD; ++k) acc += a[k] * b[k];
    Dots[idx] = acc;
}

__global__ void zero_kernel(float* __restrict__ p, int n) {
    const int idx = blockIdx.x * blockDim.x + threadIdx.x;
    if (idx < n) p[idx] = 0.f;
}

__global__ void dv_kernel(const float* __restrict__ E_sum,
                          const float* __restrict__ W,
                          const float* __restrict__ b,
                          float* __restrict__ dV) {
    const int idx = blockIdx.x * blockDim.x + threadIdx.x;
    if (idx >= NN * HD) return;
    const int i = idx / HD, c = idx - i * HD;
    float acc = (float)NN * b[c];                   // bias summed over N edges
    const float* es = E_sum + i * HD;
    for (int k = 0; k < HD; ++k) acc += es[k] * W[k * HD + c];
    dV[idx] = acc;
}

__global__ void hupd_kernel(float* __restrict__ h,
                            const float* __restrict__ dV,
                            const float* __restrict__ W,
                            const float* __restrict__ b) {
    const int idx = blockIdx.x * blockDim.x + threadIdx.x;
    if (idx >= NN * HD) return;
    const int i = idx / HD, c = idx - i * HD;
    float acc = b[c];
    const float* dv = dV + i * HD;
    for (int k = 0; k < HD; ++k) acc += dv[k] * W[k * HD + c];
    h[idx] += acc;
}

// ---------------------------------------------------------------------------
extern "C" void kernel_launch(void* const* d_in, const int* in_sizes, int n_in,
                              void* d_out, int out_size, void* d_ws, size_t ws_size,
                              hipStream_t stream) {
    const float* node_feats = (const float*)d_in[0];
    const float* edge_attr  = (const float*)d_in[1];
    const float* W_node = (const float*)d_in[2];
    const float* b_node = (const float*)d_in[3];
    const float* W_edge = (const float*)d_in[4];
    const float* b_edge = (const float*)d_in[5];
    const float* W_eij  = (const float*)d_in[6];
    const float* b_eij  = (const float*)d_in[7];
    const float* W_i    = (const float*)d_in[8];
    const float* W_j    = (const float*)d_in[9];
    const float* W_de   = (const float*)d_in[10];
    const float* b_de   = (const float*)d_in[11];
    const float* W_dv1  = (const float*)d_in[12];
    const float* b_dv1  = (const float*)d_in[13];
    const float* W_dv2  = (const float*)d_in[14];
    const float* b_dv2  = (const float*)d_in[15];

    // Output tuple (h, e) laid out flat: h then e. Compute both in place.
    float* h = (float*)d_out;
    float* e = h + (size_t)NN * HD;

    // Workspace: F_i | F_j | Dots | E_sum | dV   (~3.6 MB fp32)
    float* F_i   = (float*)d_ws;
    float* F_j   = F_i + (size_t)NN * HD;
    float* Dots  = F_j + (size_t)NN * HD;
    float* E_sum = Dots + (size_t)NEDGE;
    float* dV    = E_sum + (size_t)NN * HD;

    const int nh = NN * HD;                       // 76800
    const int bh = (nh + 255) / 256;              // 300

    init_h_kernel<<<bh, 256, 0, stream>>>(node_feats, W_node, b_node, h);
    init_e_kernel<<<(NEDGE * HD + 255) / 256, 256, 0, stream>>>(edge_attr, W_edge, b_edge, e);

    for (int it = 0; it < 3; ++it) {
        fifj_kernel<<<bh, 256, 0, stream>>>(h, W_i, W_j, F_i, F_j);
        dots_kernel<<<(NEDGE + 255) / 256, 256, 0, stream>>>(F_i, F_j, Dots);
        zero_kernel<<<bh, 256, 0, stream>>>(E_sum, nh);
        edge_update_kernel<<<NEDGE / (WPB * 16), WPB * 32, 0, stream>>>(
            e, W_eij, b_eij, W_de, b_de, Dots, E_sum);
        dv_kernel<<<bh, 256, 0, stream>>>(E_sum, W_dv1, b_dv1, dV);
        hupd_kernel<<<bh, 256, 0, stream>>>(h, dV, W_dv2, b_dv2);
    }
}
